// PDEAutoEncoder_57131654971748
// MI455X (gfx1250) — matrix operand order repaired and checked by reading, not compiled, not run on hardware
//
#include <hip/hip_runtime.h>
#include <hip/hip_bf16.h>
#include <stdint.h>

typedef unsigned short u16;
typedef __attribute__((ext_vector_type(16))) __bf16 v16bf;
typedef __attribute__((ext_vector_type(8)))  __bf16 v8bf;
typedef __attribute__((ext_vector_type(8)))  float  v8f;

// ---------- bf16 helpers (bit-level, round-to-nearest-even) ----------
__device__ __forceinline__ u16 f2bf(float f) {
    union { float f; unsigned u; } c; c.f = f;
    unsigned u = c.u;
    return (u16)((u + 0x7fffu + ((u >> 16) & 1u)) >> 16);
}
__device__ __forceinline__ float bf2f(u16 v) {
    union { unsigned u; float f; } c; c.u = ((unsigned)v) << 16;
    return c.f;
}

// ---------- WMMA fragment loads from LDS (per CDNA5 ISA 7.12.2 layouts) ----
// A 16x32 bf16: lane l (row M=l&15) holds K in {base..base+7} U {base+16..base+23},
// base = (l<16 ? 0 : 8). Two contiguous 8-elem runs.
__device__ __forceinline__ v16bf ld_afrag(const u16* row, int lane) {
    const __bf16* p = (const __bf16*)row + ((lane & 16) ? 8 : 0);
    v8bf lo = *(const v8bf*)p;
    v8bf hi = *(const v8bf*)(p + 16);
    return __builtin_shufflevector(lo, hi, 0,1,2,3,4,5,6,7,8,9,10,11,12,13,14,15);
}
// B 32x16 bf16: lane column N=l&15; K in {base..base+7} U {base+16..base+23}, base = (l<16 ? 0 : 8); row stride ld.
__device__ __forceinline__ v16bf ld_bfrag(const u16* p0, int ld, int lane) {
    const __bf16* p = (const __bf16*)p0;
    int base = (lane & 16) ? 8 : 0;
    v16bf r;
    for (int i = 0; i < 8; ++i) { r[i] = p[(base + i) * ld]; r[8 + i] = p[(base + 16 + i) * ld]; }
    return r;
}

// ======================================================================
// Encoder conv1: 1 -> 8 channels, dil=1, fp32 in, bf16 out. Small; scalar.
// ======================================================================
__global__ __launch_bounds__(256) void conv1_k(
    const float* __restrict__ x, const float* __restrict__ w,
    const float* __restrict__ bias, u16* __restrict__ h1)
{
    int idx = blockIdx.x * 256 + threadIdx.x;
    if (idx >= 4 * 62 * 4096) return;
    int pos = idx & 4095;
    int tb = idx >> 12;
    int t = tb % 62, b = tb / 62;
    int hh = pos >> 6, ww = pos & 63;
    float patch[27];
    int pi = 0;
    for (int kt = 0; kt < 3; ++kt)
        for (int kh = 0; kh < 3; ++kh)
            for (int kw = 0; kw < 3; ++kw)
                patch[pi++] = x[(b * 64 + (t + kt)) * 4096 +
                                (((hh + kh - 1) & 63) << 6) + ((ww + kw - 1) & 63)];
    for (int co = 0; co < 8; ++co) {
        float a = bias[co];
        const float* wc = w + co * 27;
        for (int k = 0; k < 27; ++k) a += wc[k] * patch[k];
        h1[((b * 8 + co) * 62 + t) * 4096 + pos] = f2bf(fmaxf(a, 0.f));
    }
}

// Convert encoder conv weights (Cout=64,Cin,3,3,3) fp32 -> A[co][Kp] bf16,
// k = tap*Cin + ci, zero padded to Kp (multiple of 32).
__global__ void convw2A(const float* __restrict__ w, u16* __restrict__ A, int CIN, int Kp)
{
    int idx = blockIdx.x * 256 + threadIdx.x;
    if (idx >= 64 * Kp) return;
    int co = idx / Kp, k = idx - co * Kp;
    u16 v = 0;
    if (k < 27 * CIN) {
        int tap = k / CIN, ci = k - tap * CIN;
        v = f2bf(w[(co * CIN + ci) * 27 + tap]);
    }
    A[idx] = v;
}

// ======================================================================
// Encoder conv layers 2..4 as WMMA GEMM.
// Block: 256 thr = 8 waves. Block tile: M=64 (full Cout) x N=64 (full W row).
// wave -> m_tile = wv&3, n_half = wv>>2; each wave keeps TWO 16x16 f32
// accumulators (N subtiles n_half*32 and n_half*32+16) and reuses its A
// fragment for both WMMAs -> 2 v_wmma per K-chunk per wave.
// K = Cin*27 padded; per 32-wide K chunk: stage A(64x32) + B(32x64) in LDS.
// ======================================================================
template<int CIN, int LOG2CIN>
__global__ __launch_bounds__(256) void enc_conv_wmma(
    const u16* __restrict__ in, int Tin,
    const u16* __restrict__ A, int Kp,
    const float* __restrict__ bias,
    u16* __restrict__ out, int Tout, int dil)
{
    __shared__ __align__(16) u16 ldsA[64 * 32];
    __shared__ __align__(16) u16 ldsB[32 * 64];
    const int tid = threadIdx.x, lane = tid & 31, wv = tid >> 5;
    const int m_tile = wv & 3, n_half = wv >> 2;
    int bx = blockIdx.x;
    int h = bx & 63;
    int rest = bx >> 6;
    int t = rest % Tout, b = rest / Tout;
    const int KTOT = 27 * CIN;
    const long inB = (long)b * CIN * Tin * 4096;
    v8f acc0 = {0.f, 0.f, 0.f, 0.f, 0.f, 0.f, 0.f, 0.f};
    v8f acc1 = {0.f, 0.f, 0.f, 0.f, 0.f, 0.f, 0.f, 0.f};
    const int nch = Kp >> 5;
    for (int kc = 0; kc < nch; ++kc) {
        {   // A chunk: 64 rows x 32 cols, 16B per thread, fully coalesced
            int row = tid >> 2, off = (tid & 3) * 8;
            *(uint4*)&ldsA[row * 32 + off] =
                *(const uint4*)(A + (long)row * Kp + kc * 32 + off);
        }
        {   // B chunk 32x64: im2col patch gather, wrap on H/W, valid on T.
            // 8 contiguous (wrapped) elems per thread, one 16B LDS store.
            int kl = tid >> 3;
            int k = kc * 32 + kl;
            int nb = (tid & 7) * 8;
            u16 vals[8];
            if (k < KTOT) {
                int ci = k & (CIN - 1);
                int tap = k >> LOG2CIN;
                int kt = tap / 9; int rem = tap - kt * 9;
                int kh = rem / 3; int kwp = rem - kh * 3;
                long base = inB + ((long)ci * Tin + (t + kt * dil)) * 4096 +
                            (long)(((h + (kh - 1) * dil) & 63) * 64);
                for (int q = 0; q < 8; ++q)
                    vals[q] = in[base + ((nb + q + (kwp - 1) * dil) & 63)];
            } else {
                for (int q = 0; q < 8; ++q) vals[q] = 0;
            }
            uint4 pk;
            pk.x = (unsigned)vals[0] | ((unsigned)vals[1] << 16);
            pk.y = (unsigned)vals[2] | ((unsigned)vals[3] << 16);
            pk.z = (unsigned)vals[4] | ((unsigned)vals[5] << 16);
            pk.w = (unsigned)vals[6] | ((unsigned)vals[7] << 16);
            *(uint4*)&ldsB[kl * 64 + nb] = pk;
        }
        __syncthreads();
        v16bf af  = ld_afrag(&ldsA[(m_tile * 16 + (lane & 15)) * 32], lane);
        v16bf bm0 = ld_bfrag(&ldsB[n_half * 32 + (lane & 15)], 64, lane);
        v16bf bm1 = ld_bfrag(&ldsB[n_half * 32 + 16 + (lane & 15)], 64, lane);
        acc0 = __builtin_amdgcn_wmma_f32_16x16x32_bf16(
            false, af, false, bm0, (short)0, acc0, false, false);
        acc1 = __builtin_amdgcn_wmma_f32_16x16x32_bf16(
            false, af, false, bm1, (short)0, acc1, false, false);
        __syncthreads();
    }
    // epilogue: bias + relu + bf16 store. D layout: VGPR r -> M=r(+8 if lane>=16)
    int co0 = m_tile * 16 + ((lane & 16) ? 8 : 0);
    int nn = n_half * 32 + (lane & 15);
    long ob = (long)b * 64 * Tout * 4096;
    for (int r = 0; r < 8; ++r) {
        int co = co0 + r;
        float bb = bias[co];
        long base = ob + ((long)co * Tout + t) * 4096 + h * 64;
        out[base + nn]      = f2bf(fmaxf(acc0[r] + bb, 0.f));
        out[base + nn + 16] = f2bf(fmaxf(acc1[r] + bb, 0.f));
    }
}

// ======================================================================
// Softmax-weighted pooling: params_v[b,p] = sum(e^{-lv} * par) / sum(e^{-lv})
// grid = 4b x 32 partial blocks, LDS tree-reduce then atomicAdd.
// ======================================================================
__global__ void zerok(float* p, int n) {
    int i = blockIdx.x * 256 + threadIdx.x;
    if (i < n) p[i] = 0.f;
}

__global__ __launch_bounds__(256) void params_reduce(
    const u16* __restrict__ h4,
    const float* __restrict__ par_w, const float* __restrict__ par_b,
    const float* __restrict__ lv_w, const float* __restrict__ lv_b,
    float* __restrict__ S)
{
    const int NS = 34 * 4096;
    int b = blockIdx.x >> 5, blk = blockIdx.x & 31;
    float s1[5] = {0, 0, 0, 0, 0}, s2[5] = {0, 0, 0, 0, 0};
    for (int i = blk * 256 + threadIdx.x; i < NS; i += 32 * 256) {
        float par[5], lv[5];
        for (int p = 0; p < 5; ++p) { par[p] = par_b[p]; lv[p] = lv_b[p]; }
        const u16* hp = h4 + (size_t)b * 64 * NS + i;
        for (int c = 0; c < 64; ++c) {
            float v = bf2f(hp[(size_t)c * NS]);
            for (int p = 0; p < 5; ++p) {
                par[p] += v * par_w[p * 64 + c];
                lv[p]  += v * lv_w[p * 64 + c];
            }
        }
        for (int p = 0; p < 5; ++p) {
            float e = __expf(-lv[p]);
            s1[p] += e; s2[p] += e * par[p];
        }
    }
    __shared__ float red[256];
    for (int p = 0; p < 5; ++p) {
        red[threadIdx.x] = s1[p]; __syncthreads();
        for (int s = 128; s > 0; s >>= 1) {
            if (threadIdx.x < s) red[threadIdx.x] += red[threadIdx.x + s];
            __syncthreads();
        }
        if (threadIdx.x == 0) atomicAdd(&S[(b * 5 + p) * 2 + 0], red[0]);
        __syncthreads();
        red[threadIdx.x] = s2[p]; __syncthreads();
        for (int s = 128; s > 0; s >>= 1) {
            if (threadIdx.x < s) red[threadIdx.x] += red[threadIdx.x + s];
            __syncthreads();
        }
        if (threadIdx.x == 0) atomicAdd(&S[(b * 5 + p) * 2 + 1], red[0]);
        __syncthreads();
    }
}

__global__ void finalize_pv(const float* __restrict__ S, float* __restrict__ pv) {
    int i = threadIdx.x;
    if (i < 20) pv[i] = S[i * 2 + 1] / S[i * 2 + 0];
}

// ======================================================================
// Small/huge hypernet MLPs: hid = relu(pv@w1+b1); out = hid@w2+b2.
// mlp_out: one thread per output column, 4 batch accumulators -> w2 read once.
// ======================================================================
__global__ __launch_bounds__(256) void mlp_hidden_k(
    const float* __restrict__ pv, const float* __restrict__ w1,
    const float* __restrict__ b1, float* __restrict__ hid, int H)
{
    int idx = blockIdx.x * 256 + threadIdx.x;
    if (idx >= 4 * H) return;
    int b = idx / H, hh = idx - b * H;
    float a = b1[hh];
    for (int k = 0; k < 5; ++k) a += pv[b * 5 + k] * w1[k * H + hh];
    hid[idx] = fmaxf(a, 0.f);
}

__global__ __launch_bounds__(256) void mlp_out_k(
    const float* __restrict__ hid, const float* __restrict__ w2,
    const float* __restrict__ b2, float* __restrict__ out, int H, int O)
{
    int o = blockIdx.x * 256 + threadIdx.x;
    if (o >= O) return;
    float a0 = 0, a1 = 0, a2 = 0, a3 = 0;
    for (int h = 0; h < H; ++h) {
        float w = w2[(size_t)h * O + o];
        __builtin_prefetch(&w2[(size_t)(h + 8) * O + o], 0, 1);
        a0 += hid[h] * w; a1 += hid[H + h] * w;
        a2 += hid[2 * H + h] * w; a3 += hid[3 * H + h] * w;
    }
    float bb = b2[o];
    out[o] = a0 + bb; out[O + o] = a1 + bb;
    out[2 * O + o] = a2 + bb; out[3 * O + o] = a3 + bb;
}

// Convert dynamic prop weights (b,3,6400) [o=k*16+co] -> A[b][j][co=16][Kp=416] bf16
__global__ void prop2A(const float* __restrict__ pw, u16* __restrict__ A)
{
    int idx = blockIdx.x * 256 + threadIdx.x;
    if (idx >= 4 * 3 * 16 * 416) return;
    int k = idx % 416; int r = idx / 416;
    int co = r & 15; int jb = r >> 4; int j = jb % 3, b = jb / 3;
    u16 v = 0;
    if (k < 400) v = f2bf(pw[b * 19200 + j * 6400 + k * 16 + co]);
    A[idx] = v;
}

// ======================================================================
// Time loop kernels
// ======================================================================
__global__ __launch_bounds__(256) void y_init_k(
    const float* __restrict__ x, float* __restrict__ y, u16* __restrict__ ybf)
{
    int idx = blockIdx.x * 256 + threadIdx.x;
    if (idx >= 4 * 4096) return;
    int b = idx >> 12, pos = idx & 4095;
    float v = x[b * 64 * 4096 + pos];   // x[b,0,t=0,:,:]
    y[idx] = v;
    ybf[idx] = f2bf(v);
}

// in conv: 1 -> 16 channels, K=25. Scalar (tiny).
__global__ __launch_bounds__(256) void in_conv_k(
    const u16* __restrict__ ybf, const float* __restrict__ inw,
    const float* __restrict__ inb, u16* __restrict__ z)
{
    int idx = blockIdx.x * 256 + threadIdx.x;
    if (idx >= 4 * 4096) return;
    int b = idx >> 12, pos = idx & 4095;
    int h = pos >> 6, w = pos & 63;
    float patch[25];
    for (int kh = 0; kh < 5; ++kh)
        for (int kw = 0; kw < 5; ++kw)
            patch[kh * 5 + kw] = bf2f(ybf[(b << 12) +
                (((h + kh - 2) & 63) << 6) + ((w + kw - 2) & 63)]);
    const float* W = inw + b * 400;   // [k=kh*5+kw][co=16]
    for (int co = 0; co < 16; ++co) {
        float a = inb[b * 16 + co];
        for (int k = 0; k < 25; ++k) a += patch[k] * W[k * 16 + co];
        z[((b * 16 + co) << 12) + pos] = f2bf(fmaxf(a, 0.f));
    }
}

// prop conv: 16 -> 16, K=400 (pad 416), WMMA. Block = 8 waves, each a 16x16
// tile: M=16 couts, N = 8x16 = 128 spatial positions. Grid = 4b x 32 ntiles.
__global__ __launch_bounds__(256) void prop_conv_wmma(
    const u16* __restrict__ zin, const u16* __restrict__ Aall,
    const float* __restrict__ pb, u16* __restrict__ zout, int j)
{
    __shared__ __align__(16) u16 ldsA[16 * 32];
    __shared__ __align__(16) u16 ldsB[32 * 128];
    const int tid = threadIdx.x, lane = tid & 31, wv = tid >> 5;
    int b = blockIdx.x >> 5;
    int n0 = (blockIdx.x & 31) * 128;
    const u16* A = Aall + (size_t)(b * 3 + j) * 16 * 416;
    const u16* zb = zin + (size_t)b * 16 * 4096;
    v8f acc = {0.f, 0.f, 0.f, 0.f, 0.f, 0.f, 0.f, 0.f};
    for (int kc = 0; kc < 13; ++kc) {
        if (tid < 128) {   // A chunk 16x32
            int row = tid >> 3, off = (tid & 7) * 4;
            *(uint2*)&ldsA[row * 32 + off] =
                *(const uint2*)(A + row * 416 + kc * 32 + off);
        }
        {   // B chunk 32x128: gather 5x5 wrap patches of z
            int kl = tid >> 3;
            int k = kc * 32 + kl;
            int nb = (tid & 7) * 16;
            u16* dst = &ldsB[kl * 128 + nb];
            if (k < 400) {
                int ci = k / 25; int r = k - ci * 25;
                int kh = r / 5; int kwp = r - kh * 5;
                const u16* src = zb + (ci << 12);
                for (int q = 0; q < 16; ++q) {
                    int n = n0 + nb + q;
                    int hin = ((n >> 6) + kh - 2) & 63;
                    int win = ((n & 63) + kwp - 2) & 63;
                    dst[q] = src[hin * 64 + win];
                }
            } else {
                for (int q = 0; q < 16; ++q) dst[q] = 0;
            }
        }
        __syncthreads();
        v16bf af = ld_afrag(&ldsA[(lane & 15) * 32], lane);
        v16bf bm = ld_bfrag(&ldsB[wv * 16 + (lane & 15)], 128, lane);
        acc = __builtin_amdgcn_wmma_f32_16x16x32_bf16(
            false, af, false, bm, (short)0, acc, false, false);
        __syncthreads();
    }
    const float* bias = pb + (b * 3 + j) * 16;
    int co0 = (lane & 16) ? 8 : 0;
    for (int r = 0; r < 8; ++r) {
        int co = co0 + r;
        float v = fmaxf(acc[r] + bias[co], 0.f);
        zout[((b * 16 + co) << 12) + n0 + wv * 16 + (lane & 15)] = f2bf(v);
    }
}

// out conv (16 -> 1) fused with tanh Euler update and output write.
__global__ __launch_bounds__(256) void out_update_k(
    const u16* __restrict__ z, const float* __restrict__ outw,
    const float* __restrict__ outb, const float* __restrict__ cutoff,
    float* __restrict__ y, u16* __restrict__ ybf,
    float* __restrict__ dout, int step, int depth)
{
    int idx = blockIdx.x * 256 + threadIdx.x;
    if (idx >= 4 * 4096) return;
    int b = idx >> 12, pos = idx & 4095;
    int h = pos >> 6, w = pos & 63;
    const float* W = outw + b * 400;   // k = ci*25 + kh*5 + kw
    float a = outb[b];
    for (int ci = 0; ci < 16; ++ci) {
        const u16* zz = z + ((b * 16 + ci) << 12);
        const float* wc = W + ci * 25;
        for (int kh = 0; kh < 5; ++kh) {
            int hin = ((h + kh - 2) & 63) << 6;
            for (int kw = 0; kw < 5; ++kw)
                a += bf2f(zz[hin + ((w + kw - 2) & 63)]) * wc[kh * 5 + kw];
        }
    }
    float c = cutoff[0];
    float yn = y[idx] + c * tanhf((1e-6f * a) / c);
    y[idx] = yn;
    ybf[idx] = f2bf(yn);
    dout[(b * depth + step) * 4096 + pos] = yn;   // (B,1,depth,H,W)
}

// ======================================================================
// Host driver
// ======================================================================
extern "C" void kernel_launch(void* const* d_in, const int* in_sizes, int n_in,
                              void* d_out, int out_size, void* d_ws, size_t ws_size,
                              hipStream_t stream)
{
    (void)in_sizes; (void)n_in; (void)ws_size;
    const float* x = (const float*)d_in[0];
    // d_in[1] = depth (device scalar) — recover on host from out_size instead
    int depth = out_size / (4 * 64 * 64);

    char* ws = (char*)d_ws;
    size_t off = 0;
    auto alloc = [&](size_t bytes) -> char* {
        char* p = ws + off;
        off = (off + bytes + 255) & ~(size_t)255;
        return p;
    };
    u16*   h1    = (u16*)alloc((size_t)4 * 8 * 62 * 4096 * 2);
    u16*   h2    = (u16*)alloc((size_t)4 * 64 * 58 * 4096 * 2);
    u16*   h3    = (u16*)alloc((size_t)4 * 64 * 50 * 4096 * 2);
    u16*   h4    = (u16*)alloc((size_t)4 * 64 * 34 * 4096 * 2);
    u16*   A2    = (u16*)alloc((size_t)64 * 224 * 2);
    u16*   A3    = (u16*)alloc((size_t)64 * 1728 * 2);
    u16*   A4    = (u16*)alloc((size_t)64 * 1728 * 2);
    float* S     = (float*)alloc(40 * 4);
    float* pv    = (float*)alloc(20 * 4);
    float* hid   = (float*)alloc((size_t)4 * 12288 * 4);
    float* inw   = (float*)alloc(1600 * 4);
    float* inbv  = (float*)alloc(64 * 4);
    float* outw  = (float*)alloc(1600 * 4);
    float* outbv = (float*)alloc(4 * 4);
    float* propw = (float*)alloc((size_t)76800 * 4);
    float* propbv= (float*)alloc(192 * 4);
    u16*   Aprop = (u16*)alloc((size_t)79872 * 2);
    float* yf    = (float*)alloc(16384 * 4);
    u16*   ybf   = (u16*)alloc(16384 * 2);
    u16*   za    = (u16*)alloc(262144 * 2);
    u16*   zbuf  = (u16*)alloc(262144 * 2);

    // ---- encoder ----
    convw2A<<<(64 * 224 + 255) / 256, 256, 0, stream>>>((const float*)d_in[4], A2, 8, 224);
    convw2A<<<(64 * 1728 + 255) / 256, 256, 0, stream>>>((const float*)d_in[6], A3, 64, 1728);
    convw2A<<<(64 * 1728 + 255) / 256, 256, 0, stream>>>((const float*)d_in[8], A4, 64, 1728);
    conv1_k<<<(4 * 62 * 4096 + 255) / 256, 256, 0, stream>>>(
        x, (const float*)d_in[2], (const float*)d_in[3], h1);
    enc_conv_wmma<8, 3><<<4 * 58 * 64, 256, 0, stream>>>(
        h1, 62, A2, 224, (const float*)d_in[5], h2, 58, 2);
    enc_conv_wmma<64, 6><<<4 * 50 * 64, 256, 0, stream>>>(
        h2, 58, A3, 1728, (const float*)d_in[7], h3, 50, 4);
    enc_conv_wmma<64, 6><<<4 * 34 * 64, 256, 0, stream>>>(
        h3, 50, A4, 1728, (const float*)d_in[9], h4, 34, 8);

    // ---- pooled latent params ----
    zerok<<<1, 64, 0, stream>>>(S, 40);
    params_reduce<<<128, 256, 0, stream>>>(
        h4, (const float*)d_in[10], (const float*)d_in[11],
        (const float*)d_in[12], (const float*)d_in[13], S);
    finalize_pv<<<1, 32, 0, stream>>>(S, pv);

    // ---- hypernet MLPs (sequential; share hid scratch) ----
    auto mlp = [&](int i0, float* outbuf, int H, int O) {
        mlp_hidden_k<<<(4 * H + 255) / 256, 256, 0, stream>>>(
            pv, (const float*)d_in[i0], (const float*)d_in[i0 + 1], hid, H);
        mlp_out_k<<<(O + 255) / 256, 256, 0, stream>>>(
            hid, (const float*)d_in[i0 + 2], (const float*)d_in[i0 + 3], outbuf, H, O);
    };
    mlp(14, inw,    256,   400);
    mlp(18, inbv,    64,    16);
    mlp(22, outw,   256,   400);
    mlp(26, outbv,    4,     1);
    mlp(30, propw, 12288, 19200);
    mlp(34, propbv,  192,    48);
    prop2A<<<(79872 + 255) / 256, 256, 0, stream>>>(propw, Aprop);

    // ---- time integration ----
    y_init_k<<<64, 256, 0, stream>>>(x, yf, ybf);
    for (int s = 0; s < depth; ++s) {
        in_conv_k<<<64, 256, 0, stream>>>(ybf, inw, inbv, za);
        prop_conv_wmma<<<128, 256, 0, stream>>>(za,   Aprop, propbv, zbuf, 0);
        prop_conv_wmma<<<128, 256, 0, stream>>>(zbuf, Aprop, propbv, za,   1);
        prop_conv_wmma<<<128, 256, 0, stream>>>(za,   Aprop, propbv, zbuf, 2);
        out_update_k<<<64, 256, 0, stream>>>(
            zbuf, outw, outbv, (const float*)d_in[38], yf, ybf,
            (float*)d_out, s, depth);
    }
}
